// AttentionHead_14353780703398
// MI455X (gfx1250) — compile-verified, matrix operand order checked by reading, not run on hardware
//
#include <hip/hip_runtime.h>

typedef _Float16 v16h __attribute__((ext_vector_type(16)));
typedef _Float16 v8h  __attribute__((ext_vector_type(8)));
typedef _Float16 v4h  __attribute__((ext_vector_type(4)));
typedef float    v8f  __attribute__((ext_vector_type(8)));

#define D_MODEL 1024
#define HEAD    64
#define SEQ     2048
#define BATCH   4

static __device__ __forceinline__ v8f wmma_f16(v16h a, v16h b, v8f c) {
    // D = A(16x32 f16) * B(32x16 f16) + C(16x16 f32)
    return __builtin_amdgcn_wmma_f32_16x16x32_f16(
        /*neg_a=*/false, a, /*neg_b=*/false, b,
        /*c_mod=*/(short)0, c, /*reuse_a=*/false, /*reuse_b=*/false);
}

static __device__ __forceinline__ v8f zero8() {
    v8f z;
#pragma unroll
    for (int i = 0; i < 8; ++i) z[i] = 0.0f;
    return z;
}

static __device__ __forceinline__ v16h combine(v8h lo, v8h hi) {
    v16h r;
#pragma unroll
    for (int i = 0; i < 8; ++i) { r[i] = lo[i]; r[8 + i] = hi[i]; }
    return r;
}

static __device__ __forceinline__ v8h ldg8(const _Float16* p) {
    return *reinterpret_cast<const v8h*>(p);   // 16B -> global_load_b128
}

// ---------------------------------------------------------------------------
// Kernel 0a: transpose weights [1024,64] f32 -> [64,1024] f16
// ---------------------------------------------------------------------------
__global__ void cvt_w_kernel(const float* __restrict__ Wq,
                             const float* __restrict__ Wk,
                             const float* __restrict__ Wv,
                             _Float16* __restrict__ Wtq,
                             _Float16* __restrict__ Wtk,
                             _Float16* __restrict__ Wtv) {
    int i = blockIdx.x * blockDim.x + threadIdx.x;     // i = c*1024 + k
    if (i >= HEAD * D_MODEL) return;
    int c = i >> 10;
    int k = i & (D_MODEL - 1);
    Wtq[i] = (_Float16)Wq[k * HEAD + c];
    Wtk[i] = (_Float16)Wk[k * HEAD + c];
    Wtv[i] = (_Float16)Wv[k * HEAD + c];
}

// ---------------------------------------------------------------------------
// Kernel 0b: x f32 -> f16, 4 elements per thread
// ---------------------------------------------------------------------------
__global__ void cvt_x_kernel(const float* __restrict__ x,
                             _Float16* __restrict__ xh, int n4) {
    int i = blockIdx.x * blockDim.x + threadIdx.x;
    if (i >= n4) return;
    float4 v = reinterpret_cast<const float4*>(x)[i];
    v4h o;
    o[0] = (_Float16)v.x; o[1] = (_Float16)v.y;
    o[2] = (_Float16)v.z; o[3] = (_Float16)v.w;
    reinterpret_cast<v4h*>(xh)[i] = o;
}

// ---------------------------------------------------------------------------
// Kernel 1: QKV projection via WMMA.
// grid.x = 512 row tiles of 16, block = 128 (4 waves), wave w -> cols [16w,16w+16)
// Q,K stored row-major [8192][64] f16; V stored transposed [B][64][2048] f16.
// ---------------------------------------------------------------------------
__global__ __launch_bounds__(128)
void proj_kernel(const _Float16* __restrict__ xh,
                 const _Float16* __restrict__ Wtq,
                 const _Float16* __restrict__ Wtk,
                 const _Float16* __restrict__ Wtv,
                 const float* __restrict__ bq,
                 const float* __restrict__ bk,
                 const float* __restrict__ bv,
                 _Float16* __restrict__ Qh,
                 _Float16* __restrict__ Kh,
                 _Float16* __restrict__ Vt) {
    const int wave = threadIdx.x >> 5;
    const int lane = threadIdx.x & 31;
    const int lm   = lane & 15;
    const int half = lane >> 4;
    const int rt   = blockIdx.x;

    const int rowA = rt * 16 + lm;        // A-matrix: lane holds row M = lane%16
    const int n    = wave * 16 + lm;      // B/C: lane holds column N = lane%16

    const _Float16* xr = xh  + (size_t)rowA * D_MODEL;
    const _Float16* wq = Wtq + (size_t)n    * D_MODEL;
    const _Float16* wk = Wtk + (size_t)n    * D_MODEL;
    const _Float16* wv = Wtv + (size_t)n    * D_MODEL;

    v8f aq = zero8(), ak = zero8(), av = zero8();
#pragma unroll 4
    for (int kb = 0; kb < D_MODEL; kb += 32) {
        const int o1 = kb + half * 8;          // K = half*8 .. +7
        const int o2 = kb + 16 + half * 8;     // K = 16+half*8 .. +7
        v16h a  = combine(ldg8(xr + o1), ldg8(xr + o2));
        v16h fq = combine(ldg8(wq + o1), ldg8(wq + o2));
        v16h fk = combine(ldg8(wk + o1), ldg8(wk + o2));
        v16h fv = combine(ldg8(wv + o1), ldg8(wv + o2));
        aq = wmma_f16(a, fq, aq);
        ak = wmma_f16(a, fk, ak);
        av = wmma_f16(a, fv, av);
    }

    const float xq = bq[n], xk = bk[n], xv = bv[n];
#pragma unroll
    for (int r = 0; r < 8; ++r) {
        const int m = rt * 16 + r + half * 8;      // C layout: VGPR r -> row
        Qh[(size_t)m * HEAD + n] = (_Float16)(aq[r] + xq);
        Kh[(size_t)m * HEAD + n] = (_Float16)(ak[r] + xk);
        const int bb = m >> 11;                     // batch
        const int s  = m & (SEQ - 1);               // seq pos
        Vt[((size_t)bb * HEAD + n) * SEQ + s] = (_Float16)(av[r] + xv);
    }
}

// ---------------------------------------------------------------------------
// Kernel 2: causal flash attention. One wave per 16-query tile, kv chunk = 64.
// grid.x = 128, block = 128 (4 waves) -> 512 tiles = 4 batches * 128 q-tiles.
// ---------------------------------------------------------------------------
__global__ __launch_bounds__(128)
void flash_kernel(const _Float16* __restrict__ Qh,
                  const _Float16* __restrict__ Kh,
                  const _Float16* __restrict__ Vt,
                  float* __restrict__ out) {
    __shared__ __align__(16) _Float16 sP[4][16][64];   // per-wave P staging (2KB/wave)

    const int wave = threadIdx.x >> 5;
    const int lane = threadIdx.x & 31;
    const int lm   = lane & 15;
    const int half = lane >> 4;

    const int tile = blockIdx.x * 4 + wave;
    const int b    = tile >> 7;
    const int q0   = (tile & 127) << 4;

    // Q fragments (A layout): lane row = q0 + lane%16, resident across kv loop
    const int rq = b * SEQ + q0 + lm;
    const _Float16* qrow = Qh + (size_t)rq * HEAD;
    const v16h aq0 = combine(ldg8(qrow +      half * 8), ldg8(qrow + 16 + half * 8));
    const v16h aq1 = combine(ldg8(qrow + 32 + half * 8), ldg8(qrow + 48 + half * 8));

    float mrun[8], lrun[8];
    v8f o[4];
#pragma unroll
    for (int r = 0; r < 8; ++r) { mrun[r] = -1e30f; lrun[r] = 0.0f; }
#pragma unroll
    for (int t = 0; t < 4; ++t) o[t] = zero8();

    const _Float16* Vtb  = Vt + (size_t)b * HEAD * SEQ;
    const float scale    = 0.125f;   // 1/sqrt(64)
    const int   kvEnd    = q0 + 16;

    for (int c0 = 0; c0 < kvEnd; c0 += 64) {
        // pull next chunk's K/V rows toward the WGP while this chunk computes
        if (c0 + 64 < kvEnd) {
            __builtin_prefetch(Kh + (size_t)(b * SEQ + c0 + 64 + lm) * HEAD, 0, 0);
            __builtin_prefetch(Vtb + (size_t)lm * SEQ + c0 + 64, 0, 0);
        }

        // ---- scores: four 16x16 tiles (kv cols c0+16t .. c0+16t+15) ----
        v8f s[4];
#pragma unroll
        for (int t = 0; t < 4; ++t) {
            const int rk = b * SEQ + c0 + t * 16 + lm;     // B: lane = kv row
            const _Float16* krow = Kh + (size_t)rk * HEAD;
            v16h bk0 = combine(ldg8(krow +      half * 8), ldg8(krow + 16 + half * 8));
            v16h bk1 = combine(ldg8(krow + 32 + half * 8), ldg8(krow + 48 + half * 8));
            v8f acc = zero8();
            acc  = wmma_f16(aq0, bk0, acc);
            s[t] = wmma_f16(aq1, bk1, acc);
        }

        // ---- causal mask + online softmax (rows at VGPR r, half-split) ----
        float p[4][8], alpha[8];
#pragma unroll
        for (int r = 0; r < 8; ++r) {
            const int mg = q0 + r + half * 8;
            float mx = -1e30f;
#pragma unroll
            for (int t = 0; t < 4; ++t) {
                const int nt = c0 + t * 16 + lm;
                const float v = (nt > mg) ? -1e30f : s[t][r] * scale;
                p[t][r] = v;
                mx = fmaxf(mx, v);
            }
#pragma unroll
            for (int off = 1; off < 16; off <<= 1)
                mx = fmaxf(mx, __shfl_xor(mx, off));
            const float mnew = fmaxf(mrun[r], mx);
            const float a    = __expf(mrun[r] - mnew);
            float sum = 0.0f;
#pragma unroll
            for (int t = 0; t < 4; ++t) {
                const float e = __expf(p[t][r] - mnew);
                p[t][r] = e;
                sum += e;
            }
#pragma unroll
            for (int off = 1; off < 16; off <<= 1)
                sum += __shfl_xor(sum, off);
            lrun[r]  = lrun[r] * a + sum;
            mrun[r]  = mnew;
            alpha[r] = a;
        }

        // rescale running output
#pragma unroll
        for (int t = 0; t < 4; ++t)
#pragma unroll
            for (int r = 0; r < 8; ++r) o[t][r] *= alpha[r];

        // ---- P: C layout -> LDS -> two A-layout fragments (16x64) ----
#pragma unroll
        for (int r = 0; r < 8; ++r) {
            const int m = r + half * 8;
#pragma unroll
            for (int t = 0; t < 4; ++t)
                sP[wave][m][t * 16 + lm] = (_Float16)p[t][r];
        }
        const v16h aP0 = combine(*reinterpret_cast<const v8h*>(&sP[wave][lm][half * 8]),
                                 *reinterpret_cast<const v8h*>(&sP[wave][lm][16 + half * 8]));
        const v16h aP1 = combine(*reinterpret_cast<const v8h*>(&sP[wave][lm][32 + half * 8]),
                                 *reinterpret_cast<const v8h*>(&sP[wave][lm][48 + half * 8]));

        // ---- O += P * V  (4 head-dim tiles x 2 kv sub-chunks) ----
#pragma unroll
        for (int ht = 0; ht < 4; ++ht) {
            const _Float16* vrow = Vtb + (size_t)(ht * 16 + lm) * SEQ + c0;
            v16h bv0 = combine(ldg8(vrow +      half * 8), ldg8(vrow + 16 + half * 8));
            v16h bv1 = combine(ldg8(vrow + 32 + half * 8), ldg8(vrow + 48 + half * 8));
            o[ht] = wmma_f16(aP0, bv0, o[ht]);
            o[ht] = wmma_f16(aP1, bv1, o[ht]);
        }
    }

    // ---- epilogue: normalize and store f32 ----
#pragma unroll
    for (int r = 0; r < 8; ++r) {
        const float inv = 1.0f / lrun[r];
        const int srow  = q0 + r + half * 8;
        float* orow = out + ((size_t)(b * SEQ + srow)) * HEAD;
#pragma unroll
        for (int ht = 0; ht < 4; ++ht)
            orow[ht * 16 + lm] = o[ht][r] * inv;
    }
}

// ---------------------------------------------------------------------------
extern "C" void kernel_launch(void* const* d_in, const int* in_sizes, int n_in,
                              void* d_out, int out_size, void* d_ws, size_t ws_size,
                              hipStream_t stream) {
    (void)in_sizes; (void)n_in; (void)out_size; (void)ws_size;
    const float* x  = (const float*)d_in[0];
    const float* Wq = (const float*)d_in[1];
    const float* bq = (const float*)d_in[2];
    const float* Wk = (const float*)d_in[3];
    const float* bk = (const float*)d_in[4];
    const float* Wv = (const float*)d_in[5];
    const float* bv = (const float*)d_in[6];
    float* out = (float*)d_out;

    char* ws = (char*)d_ws;
    size_t off = 0;
    _Float16* xh  = (_Float16*)(ws + off); off += (size_t)BATCH * SEQ * D_MODEL * 2; // 16 MB
    _Float16* Wtq = (_Float16*)(ws + off); off += (size_t)HEAD * D_MODEL * 2;
    _Float16* Wtk = (_Float16*)(ws + off); off += (size_t)HEAD * D_MODEL * 2;
    _Float16* Wtv = (_Float16*)(ws + off); off += (size_t)HEAD * D_MODEL * 2;
    _Float16* Qh  = (_Float16*)(ws + off); off += (size_t)BATCH * SEQ * HEAD * 2;
    _Float16* Kh  = (_Float16*)(ws + off); off += (size_t)BATCH * SEQ * HEAD * 2;
    _Float16* Vt  = (_Float16*)(ws + off); off += (size_t)BATCH * SEQ * HEAD * 2;

    const int nW  = HEAD * D_MODEL;                 // 65536
    const int n4  = (BATCH * SEQ * D_MODEL) / 4;    // 2097152

    cvt_w_kernel<<<(nW + 255) / 256, 256, 0, stream>>>(Wq, Wk, Wv, Wtq, Wtk, Wtv);
    cvt_x_kernel<<<(n4 + 255) / 256, 256, 0, stream>>>(x, xh, n4);
    proj_kernel<<<(BATCH * SEQ) / 16, 128, 0, stream>>>(xh, Wtq, Wtk, Wtv,
                                                        bq, bk, bv, Qh, Kh, Vt);
    flash_kernel<<<(BATCH * SEQ) / 64, 128, 0, stream>>>(Qh, Kh, Vt, out);
}
